// WindowAttention_68204080661252
// MI455X (gfx1250) — matrix operand order repaired and checked by reading, not compiled, not run on hardware
//
#include <hip/hip_runtime.h>
#include <hip/hip_bf16.h>

typedef __attribute__((ext_vector_type(16))) _Float16 v16h;
typedef __attribute__((ext_vector_type(8)))  _Float16 v8h;
typedef __attribute__((ext_vector_type(8)))  float    v8f;
typedef __attribute__((ext_vector_type(4)))  unsigned int u32x4;
typedef __attribute__((ext_vector_type(8)))  int      i32x8;
typedef __attribute__((ext_vector_type(4)))  int      i32x4;

#define N_TOK 64
#define CH    256
#define HD    32
#define NW    1024
#define XS    272   /* LDS row stride in halves for 256-wide rows (+16 pad) */
#define VS    80    /* LDS row stride in halves for 64-wide rows  (+16 pad) */
/* byte offset of the Ps region (overlaid by the fp32 x staging buffer) */
#define XF32_LDS_OFF ((3 * N_TOK * XS + CH * VS) * 2)

// ---- WMMA fragment loaders -------------------------------------------------
// A-matrix 16x32 f16 layout (ISA 7.12.2): lane l holds row m=l%16,
// K = sel*8 + {0..7} in v0..3 and K = 16 + sel*8 + {0..7} in v4..7 (sel=l/16).
__device__ __forceinline__ v16h afrag(const _Float16* rowp, int k0, int sel) {
  const _Float16* p = rowp + k0 + sel * 8;
  v8h lo = *(const v8h*)(p);
  v8h hi = *(const v8h*)(p + 16);
  return __builtin_shufflevector(lo, hi, 0,1,2,3,4,5,6,7,8,9,10,11,12,13,14,15);
}
// B-matrix 32x16 f16 layout: lane l holds column n=l%16,
// K = sel*8 + {0..7} in v0..3 and K = 16 + sel*8 + {0..7} in v4..7 (matches A).
__device__ __forceinline__ v16h bfrag(const _Float16* rowp, int k0, int sel) {
  const _Float16* p = rowp + k0 + sel * 8;
  v8h lo = *(const v8h*)(p);
  v8h hi = *(const v8h*)(p + 16);
  return __builtin_shufflevector(lo, hi, 0,1,2,3,4,5,6,7,8,9,10,11,12,13,14,15);
}
__device__ __forceinline__ v8f wmma_f16(v16h a, v16h b, v8f c) {
  return __builtin_amdgcn_wmma_f32_16x16x32_f16(false, a, false, b, (short)0, c,
                                                false, false);
}

// ---- Prep: f32->f16 transposed weights + gathered relative-position bias ---
__global__ __launch_bounds__(256) void swin_prep_kernel(
    const float* __restrict__ Wq, const float* __restrict__ Wkv,
    const float* __restrict__ Wp, const float* __restrict__ rpb,
    const int* __restrict__ rel, _Float16* __restrict__ WqT,
    _Float16* __restrict__ WkvT, _Float16* __restrict__ WpT,
    float* __restrict__ Bias) {
  int i = blockIdx.x * blockDim.x + threadIdx.x;
  if (i < 256 * 256) {  // WqT[n][k] = Wq[k][n]
    int n = i >> 8, k = i & 255;
    WqT[n * 256 + k] = (_Float16)Wq[k * 256 + n];
    WpT[n * 256 + k] = (_Float16)Wp[k * 256 + n];
  }
  if (i < 512 * 256) {  // WkvT[n][k] = Wkv[k][n], n in [0,512)
    int n = i >> 8, k = i & 255;
    WkvT[n * 256 + k] = (_Float16)Wkv[k * 512 + n];
  }
  if (i < 8 * 64 * 64) {  // Bias[h][i][j] = rpb_table[rel_idx[i][j]][h]
    int h = i >> 12, ij = i & 4095;
    Bias[i] = rpb[rel[ij] * 8 + h];
  }
}

// ---- Main fused window-attention kernel ------------------------------------
__global__ __launch_bounds__(256) void swin_attn_kernel(
    const float* __restrict__ x, const float* __restrict__ mask,
    const float* __restrict__ bq, const float* __restrict__ bkv,
    const float* __restrict__ bp, const _Float16* __restrict__ WqT,
    const _Float16* __restrict__ WkvT, const _Float16* __restrict__ WpT,
    const float* __restrict__ Bias, float* __restrict__ out) {
  extern __shared__ _Float16 smem[];
  _Float16* Xs = smem;                 // [64][XS]  x tile (f16), reused as X2
  _Float16* Qs = Xs + N_TOK * XS;      // [64][XS]  Q (scale folded in)
  _Float16* Ks = Qs + N_TOK * XS;      // [64][XS]  K row-major
  _Float16* Vt = Ks + N_TOK * XS;      // [256][VS] V transposed (channel-major)
  _Float16* Ps = Vt + CH * VS;         // [8][64][VS] softmax probs per head
  _Float16* X2 = Xs;                   // attention output (f16), overlays Xs
  float*    Xf = (float*)Ps;           // fp32 x staging (TDM dest), overlays Ps

  const int tid  = threadIdx.x;
  const int lane = tid & 31;
  const int w    = __builtin_amdgcn_readfirstlane(tid >> 5);  // uniform wave id
  const int ln   = lane & 15;
  const int sel  = lane >> 4;
  const int b    = blockIdx.x;
  const int widx = b & (NW - 1);

  const float* xb = x + (size_t)b * (N_TOK * CH);
  const float* Mw = mask + (size_t)widx * 4096;

  // -------- Phase 0: TDM DMA of the fp32 x tile into LDS (wave 0 only) ------
  if (w == 0) {
    unsigned long long ga = (unsigned long long)(const void*)xb;
    u32x4 g0;
    g0[0] = 1u;                                   // count=1 (valid descriptor)
    g0[1] = (unsigned int)XF32_LDS_OFF;           // lds_addr (bytes)
    g0[2] = (unsigned int)(ga & 0xFFFFFFFFu);     // global_addr[31:0]
    g0[3] = ((unsigned int)(ga >> 32) & 0x1FFFFFFu) | (2u << 30);  // type=2
    i32x8 g1 = {0, 0, 0, 0, 0, 0, 0, 0};
    g1[0] = (2 << 16);                  // data_size = 4 bytes
    g1[1] = (int)(16384u << 16);        // tensor_dim0 = 16384 (low 16 bits)
    g1[2] = (1 << 16);                  // tensor_dim1 = 1
    g1[3] = (int)(16384u << 16);        // tile_dim0 = 16384 elements (1 row)
    g1[5] = 16384;                      // tensor_dim0_stride
    i32x4 gz = {0, 0, 0, 0};
#if defined(__clang_major__) && __clang_major__ >= 23
    i32x8 gz8 = {0, 0, 0, 0, 0, 0, 0, 0};
    __builtin_amdgcn_tensor_load_to_lds(g0, g1, gz, gz, gz8, 0);
#else
    __builtin_amdgcn_tensor_load_to_lds(g0, g1, gz, gz, 0);
#endif
    __builtin_amdgcn_s_wait_tensorcnt(0);
  }
  // prefetch this window's mask (softmax phase) into cache: 64B per thread
  __builtin_prefetch(Mw + (tid << 4), 0, 1);
  __syncthreads();

  // -------- Phase 1: convert staged fp32 tile -> f16 working tile --------
  for (int i = tid; i < N_TOK * CH; i += 256) {
    int m = i >> 8, c = i & 255;
    Xs[m * XS + c] = (_Float16)Xf[i];
  }
  __syncthreads();

  const float scale = 0.17677669529663687f;  // 1/sqrt(32)

  // -------- Phase 2: fused QKV projection (48 col-tiles, 6 per wave) --------
  for (int mt = 0; mt < 4; ++mt) {
    const _Float16* arow = Xs + (mt * 16 + ln) * XS;
    v16h a[8];
#pragma unroll
    for (int k8 = 0; k8 < 8; ++k8) a[k8] = afrag(arow, k8 * 32, sel);
#pragma unroll
    for (int i = 0; i < 6; ++i) {
      int ct = w * 6 + i;  // scalar (w is readfirstlane-uniform)
      const _Float16* brow;
      float bb;
      if (ct < 16) {
        brow = WqT + (ct * 16 + ln) * 256;
        bb = bq[ct * 16 + ln];               // hoisted: overlaps WMMA loop
      } else {
        brow = WkvT + ((ct - 16) * 16 + ln) * 256;
        bb = bkv[(ct - 16) * 16 + ln];
      }
      v8f acc = {};
#pragma unroll
      for (int k8 = 0; k8 < 8; ++k8)
        acc = wmma_f16(a[k8], bfrag(brow, k8 * 32, sel), acc);
      if (ct < 16) {                    // Q columns: add bias, fold scale
        int cq = ct * 16 + ln;
#pragma unroll
        for (int r = 0; r < 8; ++r) {
          int m = mt * 16 + r + 8 * sel;
          Qs[m * XS + cq] = (_Float16)((acc[r] + bb) * scale);
        }
      } else if (ct < 32) {             // K half, row-major
        int c2 = (ct - 16) * 16 + ln;
#pragma unroll
        for (int r = 0; r < 8; ++r) {
          int m = mt * 16 + r + 8 * sel;
          Ks[m * XS + c2] = (_Float16)(acc[r] + bb);
        }
      } else {                          // V half, store transposed
        int c2 = (ct - 16) * 16 + ln;
#pragma unroll
        for (int r = 0; r < 8; ++r) {
          int m = mt * 16 + r + 8 * sel;
          Vt[(c2 - 256) * VS + m] = (_Float16)(acc[r] + bb);
        }
      }
    }
  }
  __syncthreads();

  // -------- Phase 3: per-head attention (wave w owns head h=w) --------
  {
    const int h = w;
    v16h qa[4], kb[4];
#pragma unroll
    for (int mt = 0; mt < 4; ++mt)
      qa[mt] = afrag(Qs + (mt * 16 + ln) * XS, h * HD, sel);
#pragma unroll
    for (int nt = 0; nt < 4; ++nt)
      kb[nt] = bfrag(Ks + (nt * 16 + ln) * XS + h * HD, 0, sel);

    v8f s[4][4];
#pragma unroll
    for (int mt = 0; mt < 4; ++mt)
#pragma unroll
      for (int nt = 0; nt < 4; ++nt) {
        v8f z = {};
        s[mt][nt] = wmma_f16(qa[mt], kb[nt], z);  // K=32 in one step
      }

    const float* Bh = Bias + h * 4096;
#pragma unroll
    for (int mt = 0; mt < 4; ++mt)
#pragma unroll
      for (int r = 0; r < 8; ++r) {
        int m = mt * 16 + r + 8 * sel;
#pragma unroll
        for (int nt = 0; nt < 4; ++nt) {
          int n = nt * 16 + ln;
          s[mt][nt][r] += Bh[m * 64 + n] + Mw[m * 64 + n];
        }
      }

    // Row softmax. Row m lives in the 16 lanes sharing sel; xor over bits 0..3.
    _Float16* Ph = Ps + h * N_TOK * VS;
#pragma unroll
    for (int mt = 0; mt < 4; ++mt) {
#pragma unroll
      for (int r = 0; r < 8; ++r) {
        float mx = s[mt][0][r];
#pragma unroll
        for (int nt = 1; nt < 4; ++nt) mx = fmaxf(mx, s[mt][nt][r]);
#pragma unroll
        for (int off = 1; off < 16; off <<= 1)
          mx = fmaxf(mx, __shfl_xor(mx, off, 32));
        float sum = 0.f;
#pragma unroll
        for (int nt = 0; nt < 4; ++nt) {
          float e = __expf(s[mt][nt][r] - mx);
          s[mt][nt][r] = e;
          sum += e;
        }
#pragma unroll
        for (int off = 1; off < 16; off <<= 1) sum += __shfl_xor(sum, off, 32);
        float inv = __builtin_amdgcn_rcpf(sum);
        int m = mt * 16 + r + 8 * sel;
#pragma unroll
        for (int nt = 0; nt < 4; ++nt)
          Ph[m * VS + nt * 16 + ln] = (_Float16)(s[mt][nt][r] * inv);
      }
    }

    // P[64,64] @ V[64,32] (two K=32 steps), accumulate f32
    v8f o[4][2] = {};
#pragma unroll
    for (int kc = 0; kc < 2; ++kc) {
      v16h pa[4], vb[2];
#pragma unroll
      for (int mt = 0; mt < 4; ++mt)
        pa[mt] = afrag(Ph + (mt * 16 + ln) * VS, kc * 32, sel);
#pragma unroll
      for (int nt = 0; nt < 2; ++nt)
        vb[nt] = bfrag(Vt + (h * HD + nt * 16 + ln) * VS, kc * 32, sel);
#pragma unroll
      for (int mt = 0; mt < 4; ++mt)
#pragma unroll
        for (int nt = 0; nt < 2; ++nt)
          o[mt][nt] = wmma_f16(pa[mt], vb[nt], o[mt][nt]);
    }
    // stash head output (f16) into X2 for the output projection
#pragma unroll
    for (int mt = 0; mt < 4; ++mt)
#pragma unroll
      for (int nt = 0; nt < 2; ++nt)
#pragma unroll
        for (int r = 0; r < 8; ++r) {
          int m = mt * 16 + r + 8 * sel;
          X2[m * XS + h * HD + nt * 16 + ln] = (_Float16)o[mt][nt][r];
        }
  }
  __syncthreads();

  // -------- Phase 4: output projection (16 col-tiles, 2 per wave) --------
  float* outb = out + (size_t)b * (N_TOK * CH);
  for (int mt = 0; mt < 4; ++mt) {
    const _Float16* arow = X2 + (mt * 16 + ln) * XS;
    v16h a[8];
#pragma unroll
    for (int k8 = 0; k8 < 8; ++k8) a[k8] = afrag(arow, k8 * 32, sel);
#pragma unroll
    for (int i = 0; i < 2; ++i) {
      int ct = w * 2 + i;
      const _Float16* brow = WpT + (ct * 16 + ln) * 256;
      int c = ct * 16 + ln;
      float bb = bp[c];                  // hoisted before WMMA loop
      v8f acc = {};
#pragma unroll
      for (int k8 = 0; k8 < 8; ++k8)
        acc = wmma_f16(a[k8], bfrag(brow, k8 * 32, sel), acc);
#pragma unroll
      for (int r = 0; r < 8; ++r) {
        int m = mt * 16 + r + 8 * sel;
        outb[m * CH + c] = acc[r] + bb;
      }
    }
  }
}

// ---- Host launcher ----------------------------------------------------------
extern "C" void kernel_launch(void* const* d_in, const int* in_sizes, int n_in,
                              void* d_out, int out_size, void* d_ws,
                              size_t ws_size, hipStream_t stream) {
  (void)in_sizes; (void)n_in; (void)out_size; (void)ws_size;
  const float* x    = (const float*)d_in[0];
  const float* mask = (const float*)d_in[1];
  const float* Wq   = (const float*)d_in[2];
  const float* bq   = (const float*)d_in[3];
  const float* Wkv  = (const float*)d_in[4];
  const float* bkv  = (const float*)d_in[5];
  const float* rpb  = (const float*)d_in[6];
  const float* Wp   = (const float*)d_in[7];
  const float* bp   = (const float*)d_in[8];
  const int*   rel  = (const int*)d_in[9];

  char* ws = (char*)d_ws;
  _Float16* WqT  = (_Float16*)(ws);                          // 128 KB
  _Float16* WkvT = (_Float16*)(ws + 131072);                 // 256 KB
  _Float16* WpT  = (_Float16*)(ws + 131072 + 262144);        // 128 KB
  float*    Bias = (float*)(ws + 131072 + 262144 + 131072);  // 128 KB

  swin_prep_kernel<<<512, 256, 0, stream>>>(Wq, Wkv, Wp, rpb, rel, WqT, WkvT,
                                            WpT, Bias);

  size_t lds_bytes =
      (size_t)(3 * N_TOK * XS + CH * VS + 8 * N_TOK * VS) * sizeof(_Float16);
  swin_attn_kernel<<<4096, 256, lds_bytes, stream>>>(
      x, mask, bq, bkv, bp, WqT, WkvT, WpT, Bias, (float*)d_out);
}
